// Spatial_Bias_Attention_16441134809595
// MI455X (gfx1250) — matrix operand order repaired and checked by reading, not compiled, not run on hardware
//
#include <hip/hip_runtime.h>
#include <hip/hip_bf16.h>

typedef __attribute__((ext_vector_type(16))) _Float16 v16h;
typedef __attribute__((ext_vector_type(8)))  _Float16 v8h;
typedef __attribute__((ext_vector_type(8)))  float    v8f;

__device__ __forceinline__ v8f wmma_f16(v16h a, v16h b, v8f c) {
  // D = A(16x32 f16) * B(32x16 f16) + C(16x16 f32)
  return __builtin_amdgcn_wmma_f32_16x16x32_f16(
      /*neg_a=*/false, a, /*neg_b=*/false, b,
      /*c_mod=*/(short)0, c, /*reuse_a=*/false, /*reuse_b=*/false);
}

// ---------------------------------------------------------------------------
// Converters: one-time (per launch) fp32 -> fp16 staging so all GEMM fragment
// loads are contiguous vector loads instead of strided scalar load+cvt chains.
// ---------------------------------------------------------------------------
__global__ __launch_bounds__(256) void cvt_x(const float* __restrict__ src,
                                             _Float16* __restrict__ dst) {
  const int i = blockIdx.x * 256 + threadIdx.x;   // grid sized exactly: 4,194,304
  dst[i] = (_Float16)src[i];
}

// Wt[c][f] = W_sel[f][c&511], c in [0,1536) selecting Wq/Wk/Wv
__global__ __launch_bounds__(256) void cvt_w_t(const float* __restrict__ Wq,
                                               const float* __restrict__ Wk,
                                               const float* __restrict__ Wv,
                                               _Float16* __restrict__ Wt) {
  const int i = blockIdx.x * 256 + threadIdx.x;   // 0..786431
  const int c = i >> 9;
  const int f = i & 511;
  const float* W = (c < 512) ? Wq : (c < 1024) ? Wk : Wv;
  Wt[(size_t)c * 512 + f] = (_Float16)W[(size_t)f * 512 + (c & 511)];
}

__global__ __launch_bounds__(256) void cvt_wout_t(const float* __restrict__ Wout,
                                                  _Float16* __restrict__ Wot) {
  const int i = blockIdx.x * 256 + threadIdx.x;   // 0..262143
  const int c = i >> 9;
  const int f = i & 511;
  Wot[(size_t)c * 512 + f] = (_Float16)Wout[(size_t)f * 512 + c];
}

// ---------------------------------------------------------------------------
// Kernel 0: pos_w = x_r . w_pos per (b,tok,h); fill bias halves of Qaug/Kaug.
// Qaug[b][h][tok][0:64]  = SCALE * q       (written by qkv_gemm)
// Qaug[b][h][tok][64:128]= pos_w * q_bias  (written here)
// Kaug[b][h][tok][0:64]  = k               (qkv_gemm)
// Kaug[b][h][tok][64:128]= k_bias          (here)
// ---------------------------------------------------------------------------
__global__ __launch_bounds__(256) void build_bias(
    const float* __restrict__ X, const float* __restrict__ qb,
    const float* __restrict__ kb, const float* __restrict__ wpos,
    _Float16* __restrict__ Qaug, _Float16* __restrict__ Kaug) {
  const int bt = blockIdx.x;            // 0..8191 = b*2048+tok
  const int h  = threadIdx.x >> 5;      // one wave per head
  const int i  = threadIdx.x & 31;
  const float* xr = X + (size_t)bt * 512 + h * 64;
  float p = xr[i] * wpos[i] + xr[i + 32] * wpos[i + 32];
#pragma unroll
  for (int d = 16; d >= 1; d >>= 1) p += __shfl_xor(p, d);
  const int b = bt >> 11, tok = bt & 2047;
  const size_t base = ((size_t)(b * 8 + h) * 2048 + tok) * 128 + 64;
  const float* qbr = qb + (size_t)bt * 64;
  const float* kbr = kb + (size_t)bt * 64;
  Qaug[base + i]      = (_Float16)(p * qbr[i]);
  Qaug[base + i + 32] = (_Float16)(p * qbr[i + 32]);
  Kaug[base + i]      = (_Float16)kbr[i];
  Kaug[base + i + 32] = (_Float16)kbr[i + 32];
}

// ---------------------------------------------------------------------------
// Kernel 1: QKV projection GEMM. C[8192,1536] = Xh[8192,512] @ Wt^T (f16).
// Wave computes 16x64 (4 accumulators, A-fragment reused across 4 B tiles).
// Fragment loads: A = 2x v8h (16B), B = 1x v16h (32B) per WMMA.
// Epilogue scatters into Qaug (x SCALE), Kaug, and transposed Vt[b][h][d][n].
// ---------------------------------------------------------------------------
__global__ __launch_bounds__(256) void qkv_gemm(
    const _Float16* __restrict__ Xh, const _Float16* __restrict__ Wt,
    _Float16* __restrict__ Qaug, _Float16* __restrict__ Kaug,
    _Float16* __restrict__ Vt) {
  const int lane  = threadIdx.x & 31;
  const int wid   = blockIdx.x * 8 + (threadIdx.x >> 5); // 0..12287
  const int mtile = wid & 511;                           // 512 M tiles
  const int ngrp  = wid >> 9;                            // 0..23 (64 cols each)
  const int m0 = mtile * 16;
  const int n0 = ngrp * 64;
  const int which = n0 >> 9;                             // 0=q,1=k,2=v (wave-uniform)
  const int lh = lane >> 4;       // half-wave selector
  const int ln = lane & 15;

  const _Float16* arow = Xh + (size_t)(m0 + ln) * 512;
  v8f acc[4] = {};
  for (int k = 0; k < 512; k += 32) {
    // A fragment (16x32 f16): row = token m0+ln; elems 0-15 at k+16*lh+e
    const _Float16* ap = arow + k + lh * 16;
    v8h lo = *(const v8h*)(ap);
    v8h hi = *(const v8h*)(ap + 8);
    v16h a;
#pragma unroll
    for (int e = 0; e < 8; ++e) { a[e] = lo[e]; a[8 + e] = hi[e]; }
#pragma unroll
    for (int nt = 0; nt < 4; ++nt) {
      // B fragment (32x16 f16): N = col n0+nt*16+ln, K = k+16*lh+e (contiguous in Wt)
      v16h bf = *(const v16h*)(Wt + (size_t)(n0 + nt * 16 + ln) * 512 + k + lh * 16);
      acc[nt] = wmma_f16(a, bf, acc[nt]);
    }
  }
#pragma unroll
  for (int nt = 0; nt < 4; ++nt) {
    const int cg = n0 + nt * 16 + ln;
    const int cw = cg & 511;
    const int h  = cw >> 6;
    const int dd = cw & 63;
#pragma unroll
    for (int r = 0; r < 8; ++r) {
      const int row = m0 + r + lh * 8;    // global token row
      const int bb  = row >> 11;
      const int tok = row & 2047;
      const int bh  = bb * 8 + h;
      const float v = acc[nt][r];
      if (which == 0)
        Qaug[((size_t)bh * 2048 + tok) * 128 + dd] = (_Float16)(v * 0.125f); // SCALE=64^-0.5
      else if (which == 1)
        Kaug[((size_t)bh * 2048 + tok) * 128 + dd] = (_Float16)v;
      else
        Vt[((size_t)bh * 64 + dd) * 2048 + tok] = (_Float16)v;
    }
  }
}

// ---------------------------------------------------------------------------
// Kernel 2: flash attention. One wave per (b,h, 16-query tile).
// Computes S^T = Kaug(16x128) x Qaug^T(128x16); the S^T D-register layout maps
// directly onto the P A-fragment layout for P(16x32) x V(32x16) — no LDS, no
// cross-lane transpose. Online softmax; m/l are per-lane (query = lane%16).
// ---------------------------------------------------------------------------
__global__ __launch_bounds__(256) void flash_attn(
    const _Float16* __restrict__ Qaug, const _Float16* __restrict__ Kaug,
    const _Float16* __restrict__ Vt, _Float16* __restrict__ Ows) {
  const int lane = threadIdx.x & 31;
  const int wid  = blockIdx.x * 8 + (threadIdx.x >> 5); // 0..4095
  const int bh = wid >> 7;                              // 0..31
  const int qt = wid & 127;                             // query tile
  const int lh = lane >> 4;
  const int ln = lane & 15;
  const _Float16* Qb = Qaug + ((size_t)bh * 2048 + qt * 16) * 128;
  const _Float16* Kb = Kaug + (size_t)bh * 2048 * 128;
  const _Float16* Vb = Vt + (size_t)bh * 64 * 2048;

  // Q^T B-fragments, register resident for the whole key loop (4 x v16h)
  v16h qf[4];
#pragma unroll
  for (int c = 0; c < 4; ++c)
    qf[c] = *(const v16h*)(Qb + (size_t)ln * 128 + c * 32 + lh * 16);

  v8f o[4] = {};
  float m = -1e30f, l = 0.f;

  for (int kc = 0; kc < 2048; kc += 32) {
    if (kc + 32 < 2048)  // prefetch next key chunk (wave-uniform condition)
      __builtin_prefetch(Kb + (size_t)(kc + 32 + ln) * 128, 0, 1);
    // S^T tiles: A = 16 keys x 32 features, B = Q^T
    v8f s[2] = {};
#pragma unroll
    for (int t = 0; t < 2; ++t) {
      const _Float16* kp = Kb + (size_t)(kc + t * 16 + ln) * 128;
#pragma unroll
      for (int c = 0; c < 4; ++c) {
        const _Float16* cp = kp + c * 32 + lh * 16;
        v8h lo = *(const v8h*)(cp);
        v8h hi = *(const v8h*)(cp + 8);
        v16h a;
#pragma unroll
        for (int e = 0; e < 8; ++e) { a[e] = lo[e]; a[8 + e] = hi[e]; }
        s[t] = wmma_f16(a, qf[c], s[t]);
      }
    }
    // Online softmax; this lane owns query `ln`, keys split with lane^16.
    float cm = s[0][0];
#pragma unroll
    for (int r = 0; r < 8; ++r) {
      cm = fmaxf(cm, s[0][r]);
      cm = fmaxf(cm, s[1][r]);
    }
    cm = fmaxf(cm, __shfl_xor(cm, 16));
    const float mnew = fmaxf(m, cm);
    const float corr = __expf(m - mnew);
    v16h pf;
    float rs = 0.f;
#pragma unroll
    for (int r = 0; r < 8; ++r) {
      const float p0 = __expf(s[0][r] - mnew);
      const float p1 = __expf(s[1][r] - mnew);
      pf[r] = (_Float16)p0;      // key rel = lh*8 + r      (A-frag elems 0-7)
      pf[8 + r] = (_Float16)p1;  // key rel = 16 + lh*8 + r (A-frag elems 8-15)
      rs += p0 + p1;
    }
    rs += __shfl_xor(rs, 16);
    l = l * corr + rs;
    m = mnew;
    // Rescale O rows (O row M = r + 8*lh; corr(q) lives in lanes q and q+16)
    float cr[8];
#pragma unroll
    for (int r = 0; r < 8; ++r) cr[r] = __shfl(corr, r + lh * 8);
#pragma unroll
    for (int ot = 0; ot < 4; ++ot)
#pragma unroll
      for (int r = 0; r < 8; ++r) o[ot][r] *= cr[r];
    // O += P(16x32) x V(32x16) per 16-dim tile; Vt rows are contiguous keys
#pragma unroll
    for (int ot = 0; ot < 4; ++ot) {
      const _Float16* vp = Vb + (size_t)(ot * 16 + ln) * 2048 + kc + lh * 8;
      v8h vlo = *(const v8h*)(vp);
      v8h vhi = *(const v8h*)(vp + 16);
      v16h vf;
#pragma unroll
      for (int e = 0; e < 8; ++e) { vf[e] = vlo[e]; vf[8 + e] = vhi[e]; }
      o[ot] = wmma_f16(pf, vf, o[ot]);
    }
  }
  // finalize: divide rows by l, store f16 to Ows[b*n, 512] (col = h*64+dd)
  float lr[8];
#pragma unroll
  for (int r = 0; r < 8; ++r) lr[r] = 1.f / __shfl(l, r + lh * 8);
  const int b = bh >> 3, h = bh & 7;
#pragma unroll
  for (int ot = 0; ot < 4; ++ot)
#pragma unroll
    for (int r = 0; r < 8; ++r) {
      const int row = b * 2048 + qt * 16 + r + lh * 8;
      Ows[(size_t)row * 512 + h * 64 + ot * 16 + ln] = (_Float16)(o[ot][r] * lr[r]);
    }
}

// ---------------------------------------------------------------------------
// Kernel 3: output projection. out[8192,512] = Ows(f16) @ Woutt^T + b_out.
// ---------------------------------------------------------------------------
__global__ __launch_bounds__(256) void out_gemm(
    const _Float16* __restrict__ Ows, const _Float16* __restrict__ Wot,
    const float* __restrict__ bout, float* __restrict__ out) {
  const int lane  = threadIdx.x & 31;
  const int wid   = blockIdx.x * 8 + (threadIdx.x >> 5); // 0..4095
  const int mtile = wid & 511;
  const int ngrp  = wid >> 9;                            // 0..7
  const int m0 = mtile * 16, n0 = ngrp * 64;
  const int lh = lane >> 4, ln = lane & 15;
  const _Float16* arow = Ows + (size_t)(m0 + ln) * 512;
  v8f acc[4] = {};
  for (int k = 0; k < 512; k += 32) {
    const _Float16* ap = arow + k + lh * 16;
    v8h lo = *(const v8h*)(ap);
    v8h hi = *(const v8h*)(ap + 8);
    v16h a;
#pragma unroll
    for (int e = 0; e < 8; ++e) { a[e] = lo[e]; a[8 + e] = hi[e]; }
#pragma unroll
    for (int nt = 0; nt < 4; ++nt) {
      v16h bf = *(const v16h*)(Wot + (size_t)(n0 + nt * 16 + ln) * 512 + k + lh * 16);
      acc[nt] = wmma_f16(a, bf, acc[nt]);
    }
  }
#pragma unroll
  for (int nt = 0; nt < 4; ++nt) {
    const int col = n0 + nt * 16 + ln;
    const float bb = bout[col];
#pragma unroll
    for (int r = 0; r < 8; ++r)
      out[(size_t)(m0 + r + lh * 8) * 512 + col] = acc[nt][r] + bb;
  }
}

// ---------------------------------------------------------------------------
// Workspace map (bytes):
//   [ 0,16M)   Qaug  f16 [b][h][n][128]
//   [16,32M)   Kaug  f16 [b][h][n][128]
//   [32,40M)   Vt    f16 [b][h][64][n]
//   [40,48M)   Ows   f16 [b*n][512]
//   [48,56M)   Xh    f16 [b*n][512]
//   [56,58M)   Wt    f16 [1536][512]   (transposed Wq|Wk|Wv)
//   [58,58.5M) Woutt f16 [512][512]    (transposed Wout)
// ---------------------------------------------------------------------------
extern "C" void kernel_launch(void* const* d_in, const int* in_sizes, int n_in,
                              void* d_out, int out_size, void* d_ws, size_t ws_size,
                              hipStream_t stream) {
  (void)in_sizes; (void)n_in; (void)out_size; (void)ws_size;
  const float* x      = (const float*)d_in[0];
  const float* q_bias = (const float*)d_in[1];
  const float* k_bias = (const float*)d_in[2];
  const float* Wq     = (const float*)d_in[3];
  const float* Wk     = (const float*)d_in[4];
  const float* Wv     = (const float*)d_in[5];
  const float* Wout   = (const float*)d_in[6];
  const float* b_out  = (const float*)d_in[7];
  const float* w_pos  = (const float*)d_in[8];
  char* ws = (char*)d_ws;
  _Float16* Qaug  = (_Float16*)(ws);
  _Float16* Kaug  = (_Float16*)(ws + ((size_t)16 << 20));
  _Float16* Vt    = (_Float16*)(ws + ((size_t)32 << 20));
  _Float16* Ows   = (_Float16*)(ws + ((size_t)40 << 20));
  _Float16* Xh    = (_Float16*)(ws + ((size_t)48 << 20));
  _Float16* Wt    = (_Float16*)(ws + ((size_t)56 << 20));
  _Float16* Woutt = (_Float16*)(ws + ((size_t)58 << 20));

  // Stage fp16 operands (contiguous / transposed layouts for WMMA fragments)
  cvt_x<<<16384, 256, 0, stream>>>(x, Xh);          // 8192*512 elements
  cvt_w_t<<<3072, 256, 0, stream>>>(Wq, Wk, Wv, Wt);
  cvt_wout_t<<<1024, 256, 0, stream>>>(Wout, Woutt);
  build_bias<<<8192, 256, 0, stream>>>(x, q_bias, k_bias, w_pos, Qaug, Kaug);
  qkv_gemm<<<1536, 256, 0, stream>>>(Xh, Wt, Qaug, Kaug, Vt);
  flash_attn<<<512, 256, 0, stream>>>(Qaug, Kaug, Vt, Ows);
  out_gemm<<<512, 256, 0, stream>>>(Ows, Woutt, b_out, (float*)d_out);
}
